// SMPL_23708219474150
// MI455X (gfx1250) — matrix operand order repaired and checked by reading, not compiled, not run on hardware
//
#include <hip/hip_runtime.h>
#include <hip/hip_bf16.h>
#include <math.h>

// ---------------------------------------------------------------------------
// SMPL forward, collapsed over the vertex dimension.
// Padded dims: R (rows of "dirs" basis) = 256 (0:template, 1..10:shapedirs,
// 11..217:posedirs, 218:ones-row for G, rest 0), V = 6912 (6890 verts),
// N = 640 (0..575: j*24+k weight products, 576..599: J_regressor cols, rest 0)
// ---------------------------------------------------------------------------

typedef __attribute__((ext_vector_type(16))) _Float16 v16h;
typedef __attribute__((ext_vector_type(8)))  float    v8f;

#define NB 1024
#define NJ 24
#define NV 6890
#define RD 256
#define VD 6912
#define ND 640

// workspace layout (all 256B aligned)
constexpr size_t SZ_APACK = 3ull * RD * VD * 2;      // f16 A_d[r][v]
constexpr size_t SZ_BT    = (size_t)ND * VD * 2;     // f16 BT[n][v]
constexpr size_t SZ_CPX   = 3ull * ND * RD * 2;      // f16 CPX_T[d][n][r]
constexpr size_t SZ_XP    = (size_t)NB * RD * 2;     // f16 x[b][r]
constexpr size_t SZ_OUT2  = 3ull * NB * ND * 4;      // f32 out2[d][b][n]
constexpr size_t SZ_RS    = (size_t)NB * NJ * 9 * 4; // f32 rotations
constexpr size_t OFF_APACK = 0;
constexpr size_t OFF_BT    = OFF_APACK + SZ_APACK;
constexpr size_t OFF_CPX   = OFF_BT + SZ_BT;
constexpr size_t OFF_XP    = OFF_CPX + SZ_CPX;
constexpr size_t OFF_OUT2  = OFF_XP + SZ_XP;
constexpr size_t OFF_RS    = OFF_OUT2 + SZ_OUT2;
constexpr size_t OFF_AFF   = OFF_RS + SZ_RS;         // f32 Aff[b][24][12]

__device__ __constant__ int g_par[NJ] = {
  -1, 0, 0, 0, 1, 2, 3, 4, 5, 6, 7, 8, 9, 9, 9, 12, 13, 14, 16, 17, 18, 19, 20, 21};

// --- CDNA5 async global->LDS copy (ASYNCcnt path), with sync fallback ------
#if __has_builtin(__builtin_amdgcn_global_load_async_to_lds_b128)
#define HAS_ASYNC_LDS 1
#else
#define HAS_ASYNC_LDS 0
#endif

#if HAS_ASYNC_LDS
// exact pointee type from the builtin's signature:
//   int __attribute__((__vector_size__(4*sizeof(int)))) __device__/__shared__ *
typedef int v4i_raw __attribute__((__vector_size__(4 * sizeof(int))));
typedef __attribute__((address_space(1))) v4i_raw glob_v4i;
typedef __attribute__((address_space(3))) v4i_raw lds_v4i;
#endif

__device__ __forceinline__ void copy16_g2l(const _Float16* __restrict__ g,
                                           _Float16* __restrict__ l) {
#if HAS_ASYNC_LDS
  // flat LDS addresses carry the LDS byte offset in the low 32 bits (ISA 10.2)
  __builtin_amdgcn_global_load_async_to_lds_b128(
      (glob_v4i*)(unsigned long long)g,
      (lds_v4i*)(unsigned int)(unsigned long long)l,
      0, 0);
#else
  *(uint4*)l = *(const uint4*)g;
#endif
}

__device__ __forceinline__ void wait_async_copies() {
#if HAS_ASYNC_LDS
#if __has_builtin(__builtin_amdgcn_s_wait_asynccnt)
  __builtin_amdgcn_s_wait_asynccnt(0);
#else
  asm volatile("s_wait_asynccnt 0x0" ::: "memory");
#endif
#endif
}

// --- pack A_d[r][v] = Dirs_r[3v+d] as f16 --------------------------------
__global__ void pack_A_kernel(const float* __restrict__ vt,
                              const float* __restrict__ sd,
                              const float* __restrict__ pd,
                              _Float16* __restrict__ Ap) {
  long idx = (long)blockIdx.x * blockDim.x + threadIdx.x;
  if (idx >= 3L * RD * VD) return;
  int v = (int)(idx % VD);
  long t = idx / VD;
  int r = (int)(t % RD);
  int d = (int)(t / RD);
  float val = 0.f;
  if (v < NV) {
    if (r == 0)        val = vt[3 * v + d];
    else if (r <= 10)  val = sd[(r - 1) * (NV * 3) + 3 * v + d];
    else if (r <= 217) val = pd[(r - 11) * (NV * 3) + 3 * v + d];
    else if (r == 218) val = 1.0f;  // ones row -> G[j,k] = sum_v Jr*w
  }
  Ap[idx] = (_Float16)val;
}

// --- pack BT[n][v]: n<576 -> Jr[v,j]*w[v,k]; 576..599 -> Jr[v,j] ----------
__global__ void pack_BT_kernel(const float* __restrict__ Jr,
                               const float* __restrict__ wgt,
                               _Float16* __restrict__ Bp) {
  long idx = (long)blockIdx.x * blockDim.x + threadIdx.x;
  if (idx >= (long)ND * VD) return;
  int v = (int)(idx % VD);
  int n = (int)(idx / VD);
  float val = 0.f;
  if (v < NV) {
    if (n < 576)      val = Jr[v * NJ + (n / NJ)] * wgt[v * NJ + (n % NJ)];
    else if (n < 600) val = Jr[v * NJ + (n - 576)];
  }
  Bp[idx] = (_Float16)val;
}

// --- Rodrigues per (b, joint); also builds x[b] = [1, beta, pose_feat, 0] --
__global__ void rodrigues_kernel(const float* __restrict__ theta,
                                 const float* __restrict__ beta,
                                 float* __restrict__ Rs,
                                 _Float16* __restrict__ xp) {
  int idx = blockIdx.x * blockDim.x + threadIdx.x;
  if (idx >= NB * NJ) return;
  int b = idx / NJ, jt = idx % NJ;
  float tx = theta[b * 72 + jt * 3 + 0];
  float ty = theta[b * 72 + jt * 3 + 1];
  float tz = theta[b * 72 + jt * 3 + 2];
  float ax = tx + 1e-8f, ay = ty + 1e-8f, az = tz + 1e-8f;
  float angle = sqrtf(ax * ax + ay * ay + az * az);
  float inva = 1.0f / angle;
  float h = 0.5f * angle;
  float w = cosf(h), s = sinf(h);
  float qx = s * tx * inva, qy = s * ty * inva, qz = s * tz * inva;
  float qn = rsqrtf(w * w + qx * qx + qy * qy + qz * qz);
  w *= qn; qx *= qn; qy *= qn; qz *= qn;
  float w2 = w * w, x2 = qx * qx, y2 = qy * qy, z2 = qz * qz;
  float wx = w * qx, wy = w * qy, wz = w * qz;
  float xy = qx * qy, xz = qx * qz, yz = qy * qz;
  float R[9];
  R[0] = w2 + x2 - y2 - z2; R[1] = 2.f * (xy - wz);     R[2] = 2.f * (xz + wy);
  R[3] = 2.f * (xy + wz);   R[4] = w2 - x2 + y2 - z2;   R[5] = 2.f * (yz - wx);
  R[6] = 2.f * (xz - wy);   R[7] = 2.f * (yz + wx);     R[8] = w2 - x2 - y2 + z2;
#pragma unroll
  for (int i = 0; i < 9; ++i) Rs[(b * NJ + jt) * 9 + i] = R[i];
  if (jt >= 1) {
#pragma unroll
    for (int i = 0; i < 9; ++i)
      xp[b * RD + 11 + (jt - 1) * 9 + i] = (_Float16)(R[i] - ((i % 4 == 0) ? 1.f : 0.f));
  } else {
    xp[b * RD + 0] = (_Float16)1.0f;
    for (int q = 0; q < 10; ++q) xp[b * RD + 1 + q] = (_Float16)beta[b * 10 + q];
    for (int r = 218; r < RD; ++r) xp[b * RD + r] = (_Float16)0.0f;  // G row excluded from x
  }
}

// --- WMMA f16 GEMM: C = A(MxK) * B^T(NxK), both K-major, f32 accumulate ----
// Block: 128 threads (4 waves), tile 64x64, K step 32, double-buffered LDS
// with async global->LDS copies overlapping the WMMA work.
// transC_f16 != 0: store f16 transposed C[n*ldC + m]; else f32 C[m*ldC + n].
__global__ __launch_bounds__(128)
void gemm_wmma_kernel(const _Float16* __restrict__ A, long sAz,
                      const _Float16* __restrict__ B, long sBz,
                      void* __restrict__ C, long sCz,
                      int K, int transC_f16, int ldC) {
  __shared__ _Float16 As[2][64 * 32];
  __shared__ _Float16 Bs[2][64 * 32];
  const int tid  = threadIdx.x;
  const int lane = tid & 31;
  const int wave = tid >> 5;
  const int wm = (wave >> 1) * 32;
  const int wn = (wave & 1) * 32;
  const int m0 = blockIdx.x * 64;
  const int n0 = blockIdx.y * 64;
  const int z  = blockIdx.z;
  A += (size_t)z * sAz;
  B += (size_t)z * sBz;

  const int lrow = lane & 15;
  const int hi   = lane >> 4;

  // staging assignment: thread owns rows r0 and r0+32, 8-half column chunk c0
  const int r0 = tid >> 2;
  const int c0 = (tid & 3) * 8;
  const _Float16* gA0 = A + (size_t)(m0 + r0) * K + c0;
  const _Float16* gA1 = A + (size_t)(m0 + r0 + 32) * K + c0;
  const _Float16* gB0 = B + (size_t)(n0 + r0) * K + c0;
  const _Float16* gB1 = B + (size_t)(n0 + r0 + 32) * K + c0;
  const int ls0 = r0 * 32 + c0;
  const int ls1 = (r0 + 32) * 32 + c0;

  auto stage = [&](int buf, int kt) {
    copy16_g2l(gA0 + kt, &As[buf][ls0]);
    copy16_g2l(gA1 + kt, &As[buf][ls1]);
    copy16_g2l(gB0 + kt, &Bs[buf][ls0]);
    copy16_g2l(gB1 + kt, &Bs[buf][ls1]);
  };

  v8f c00 = {}, c01 = {}, c10 = {}, c11 = {};
  union V16 { v16h v; uint4 u[2]; };

  const int nIt = K >> 5;
  stage(0, 0);

  for (int it = 0; it < nIt; ++it) {
    const int cur = it & 1;
    wait_async_copies();   // this wave's copies into `cur` have landed
    __syncthreads();       // ... and everyone's; also retires readers of 1-cur
    if (it + 1 < nIt) stage(1 - cur, (it + 1) * 32);  // overlap with compute

    // A frag (16x32 f16): lane m = lrow; halves = k[ka..ka+15] contiguous
    const int ka = hi * 16;
    V16 a0t, a1t;
    {
      const _Float16* pa = &As[cur][(wm + lrow) * 32 + ka];
      a0t.u[0] = *(const uint4*)pa;
      a0t.u[1] = *(const uint4*)(pa + 8);
      pa = &As[cur][(wm + 16 + lrow) * 32 + ka];
      a1t.u[0] = *(const uint4*)pa;
      a1t.u[1] = *(const uint4*)(pa + 8);
    }
    // B frag (32x16 f16): lane n = lrow; halves = k[kb..kb+15] contiguous
    const int kb = hi * 16;
    v16h b0 = *(const v16h*)&Bs[cur][(wn + lrow) * 32 + kb];
    v16h b1 = *(const v16h*)&Bs[cur][(wn + 16 + lrow) * 32 + kb];

    c00 = __builtin_amdgcn_wmma_f32_16x16x32_f16(false, a0t.v, false, b0, (short)0, c00, false, false);
    c01 = __builtin_amdgcn_wmma_f32_16x16x32_f16(false, a0t.v, false, b1, (short)0, c01, false, false);
    c10 = __builtin_amdgcn_wmma_f32_16x16x32_f16(false, a1t.v, false, b0, (short)0, c10, false, false);
    c11 = __builtin_amdgcn_wmma_f32_16x16x32_f16(false, a1t.v, false, b1, (short)0, c11, false, false);
  }

  auto storeFrag = [&](v8f cc, int mf, int nf) {
    int n = nf + lrow;
    int mrow = mf + hi * 8;   // VGPR r holds M = mrow + r
    if (transC_f16) {
      _Float16* Ch = (_Float16*)C + (size_t)z * sCz;
      alignas(16) _Float16 tmp[8];
#pragma unroll
      for (int r = 0; r < 8; ++r) tmp[r] = (_Float16)cc[r];
      *(uint4*)&Ch[(size_t)n * ldC + mrow] = *(const uint4*)tmp;
    } else {
      float* Cf = (float*)C + (size_t)z * sCz;
#pragma unroll
      for (int r = 0; r < 8; ++r) Cf[(size_t)(mrow + r) * ldC + n] = cc[r];
    }
  };
  storeFrag(c00, m0 + wm,      n0 + wn);
  storeFrag(c01, m0 + wm,      n0 + wn + 16);
  storeFrag(c10, m0 + wm + 16, n0 + wn);
  storeFrag(c11, m0 + wm + 16, n0 + wn + 16);
}

// --- zero posedirs rows in the J-regressor columns (J uses v_shaped only) --
__global__ void fixup_kernel(_Float16* __restrict__ cpx) {
  int idx = blockIdx.x * blockDim.x + threadIdx.x;
  if (idx >= 3 * NJ * 207) return;
  int r = 11 + idx % 207;
  int t = idx / 207;
  int j = t % NJ;
  int d = t / NJ;
  cpx[(size_t)d * ND * RD + (size_t)(576 + j) * RD + r] = (_Float16)0.0f;
}

// --- kinematic chain: Aff[b][k] = (Rglob_k | t_k - Rglob_k * J_k) ----------
__global__ void chain_kernel(const float* __restrict__ out2,
                             const float* __restrict__ Rs,
                             float* __restrict__ Aff) {
  int b = blockIdx.x * blockDim.x + threadIdx.x;
  if (b >= NB) return;
  const long PS = (long)NB * ND;
  float J[NJ][3];
#pragma unroll
  for (int j = 0; j < NJ; ++j)
#pragma unroll
    for (int c = 0; c < 3; ++c)
      J[j][c] = out2[(long)c * PS + (long)b * ND + 576 + j];

  float* Ab = Aff + (size_t)b * (NJ * 12);
  const float* Rb = Rs + (size_t)b * (NJ * 9);
  // root
#pragma unroll
  for (int c = 0; c < 3; ++c) {
    Ab[c * 4 + 0] = Rb[c * 3 + 0];
    Ab[c * 4 + 1] = Rb[c * 3 + 1];
    Ab[c * 4 + 2] = Rb[c * 3 + 2];
    Ab[c * 4 + 3] = J[0][c];
  }
  for (int i = 1; i < NJ; ++i) {
    int p = g_par[i];
    float* Ai = Ab + i * 12;
    const float* Apar = Ab + p * 12;
    const float* Ri = Rb + i * 9;
    float tl0 = J[i][0] - J[p][0];
    float tl1 = J[i][1] - J[p][1];
    float tl2 = J[i][2] - J[p][2];
#pragma unroll
    for (int c = 0; c < 3; ++c) {
      float p0 = Apar[c * 4 + 0], p1 = Apar[c * 4 + 1], p2 = Apar[c * 4 + 2];
#pragma unroll
      for (int e = 0; e < 3; ++e)
        Ai[c * 4 + e] = p0 * Ri[0 * 3 + e] + p1 * Ri[1 * 3 + e] + p2 * Ri[2 * 3 + e];
      Ai[c * 4 + 3] = p0 * tl0 + p1 * tl1 + p2 * tl2 + Apar[c * 4 + 3];
    }
  }
  // A = results - results*[J;0] in last column
  for (int k = 0; k < NJ; ++k) {
    float* Ak = Ab + k * 12;
#pragma unroll
    for (int c = 0; c < 3; ++c)
      Ak[c * 4 + 3] -= Ak[c * 4 + 0] * J[k][0] + Ak[c * 4 + 1] * J[k][1] + Ak[c * 4 + 2] * J[k][2];
  }
}

// --- joints[b,j,c] = sum_k ( A[b,k,c,:3].M[b,jk,:] + A[b,k,c,3]*G[j,k] ) ---
__global__ void final_kernel(const float* __restrict__ out2,
                             const float* __restrict__ Aff,
                             const _Float16* __restrict__ cpx,
                             float* __restrict__ out) {
  int idx = blockIdx.x * blockDim.x + threadIdx.x;
  if (idx >= NB * NJ * 3) return;
  int b = idx / 72;
  int rem = idx % 72;
  int j = rem / 3;
  int c = rem % 3;
  const long PS = (long)NB * ND;
  const float* Ab = Aff + (size_t)b * (NJ * 12);
  float acc = 0.f;
#pragma unroll 4
  for (int k = 0; k < NJ; ++k) {
    int n = j * NJ + k;
    const float* ar = Ab + k * 12 + c * 4;
    float m0 = out2[0 * PS + (long)b * ND + n];
    float m1 = out2[1 * PS + (long)b * ND + n];
    float m2 = out2[2 * PS + (long)b * ND + n];
    float g  = (float)cpx[(size_t)n * RD + 218];  // plane d=0, ones-row
    acc += ar[0] * m0 + ar[1] * m1 + ar[2] * m2 + ar[3] * g;
  }
  out[idx] = acc;
}

// ---------------------------------------------------------------------------
extern "C" void kernel_launch(void* const* d_in, const int* in_sizes, int n_in,
                              void* d_out, int out_size, void* d_ws, size_t ws_size,
                              hipStream_t stream) {
  (void)in_sizes; (void)n_in; (void)out_size; (void)ws_size;
  const float* beta  = (const float*)d_in[0];
  const float* theta = (const float*)d_in[1];
  const float* vt    = (const float*)d_in[2];
  const float* sd    = (const float*)d_in[3];
  const float* pd    = (const float*)d_in[4];
  const float* Jr    = (const float*)d_in[5];
  const float* wgt   = (const float*)d_in[6];

  char* ws = (char*)d_ws;
  _Float16* Ap   = (_Float16*)(ws + OFF_APACK);
  _Float16* Bp   = (_Float16*)(ws + OFF_BT);
  _Float16* Cpx  = (_Float16*)(ws + OFF_CPX);
  _Float16* Xp   = (_Float16*)(ws + OFF_XP);
  float*    Out2 = (float*)(ws + OFF_OUT2);
  float*    Rsb  = (float*)(ws + OFF_RS);
  float*    Aff  = (float*)(ws + OFF_AFF);

  // independent prep
  pack_A_kernel<<<(int)((3L * RD * VD + 255) / 256), 256, 0, stream>>>(vt, sd, pd, Ap);
  pack_BT_kernel<<<(int)(((long)ND * VD + 255) / 256), 256, 0, stream>>>(Jr, wgt, Bp);
  rodrigues_kernel<<<(NB * NJ + 255) / 256, 256, 0, stream>>>(theta, beta, Rsb, Xp);

  // GEMM1: CPX_T[d][n][r] = sum_v A_d[r,v] * BT[n,v]   (M=256, N=640, K=6912)
  gemm_wmma_kernel<<<dim3(RD / 64, ND / 64, 3), 128, 0, stream>>>(
      Ap, (long)RD * VD, Bp, 0L, (void*)Cpx, (long)ND * RD, VD, 1, RD);

  fixup_kernel<<<(3 * NJ * 207 + 255) / 256, 256, 0, stream>>>(Cpx);

  // GEMM2: out2[d][b][n] = sum_r x[b,r] * CPX_T[d][n,r]  (M=1024, N=640, K=256)
  gemm_wmma_kernel<<<dim3(NB / 64, ND / 64, 3), 128, 0, stream>>>(
      Xp, 0L, Cpx, (long)ND * RD, (void*)Out2, (long)NB * ND, RD, 0, ND);

  chain_kernel<<<NB / 128, 128, 0, stream>>>(Out2, Rsb, Aff);
  final_kernel<<<(NB * NJ * 3 + 255) / 256, 256, 0, stream>>>(Out2, Aff, Cpx, (float*)d_out);
}
